// VAE_regression_59450937311853
// MI455X (gfx1250) — compile-verified, matrix operand order checked by reading, not run on hardware
//
#include <hip/hip_runtime.h>
#include <hip/hip_bf16.h>
#include <string.h>

// ---------------- problem dims ----------------
#define Bn 2048
#define Dn 4096
#define Hn 2048
#define Ln 512
#define Cn 10

typedef __attribute__((ext_vector_type(16))) __bf16 v16bf;
typedef __attribute__((ext_vector_type(8)))  float  v8f;
typedef __attribute__((ext_vector_type(4)))  int    v4i;

// CDNA5 async global->LDS DMA path (ASYNCcnt-tracked), if the toolchain has it.
#if defined(__AMDGCN__) && __has_builtin(__builtin_amdgcn_global_load_async_to_lds_b128)
#define USE_ASYNC_LDS 1
// builtin wants: (v4i addrspace(1)*, v4i addrspace(3)*, imm offset, imm cpol)
typedef __attribute__((address_space(1))) v4i v4i_as1;
typedef __attribute__((address_space(3))) v4i v4i_as3;
#else
#define USE_ASYNC_LDS 0
#endif

__device__ __forceinline__ void wait_async_lds() {
#if USE_ASYNC_LDS
#if __has_builtin(__builtin_amdgcn_s_wait_asynccnt)
    __builtin_amdgcn_s_wait_asynccnt(0);
#else
    asm volatile("s_wait_asynccnt 0x0" ::: "memory");
#endif
#endif
}

__device__ __forceinline__ __bf16 f2bf(float f) {
    unsigned u = __float_as_uint(f);
    unsigned r = (u + 0x7FFFu + ((u >> 16) & 1u)) >> 16;
    unsigned short s = (unsigned short)r;
    return __builtin_bit_cast(__bf16, s);
}

// ---------------- f32 -> bf16 conversion ----------------
__global__ void cvt_bf16_kernel(const float* __restrict__ src,
                                __bf16* __restrict__ dst, int n) {
    int i = blockIdx.x * blockDim.x + threadIdx.x;
    if (i < n) dst[i] = f2bf(src[i]);
}

// ---------------- tiled bf16 WMMA GEMM ----------------
// out[m,n] = sum_k A[m,k] * W[n,k]  (+bias[n]) (ReLU optional)
// A: [M,K] bf16 row-major; W: [N,K] bf16 row-major (torch Linear weight).
// Block: 256 threads (8 waves). Block tile 128(M) x 128(N), K-step 32.
// Wave grid 4(M) x 2(N); each wave -> 32x64 via eight 16x16 WMMA accums.
// Double-buffered LDS staging via GLOBAL_LOAD_ASYNC_TO_LDS_B128 when available.
__global__ void gemm_bf16_kernel(const __bf16* __restrict__ A,
                                 const __bf16* __restrict__ W,
                                 const float* __restrict__ bias,
                                 float* __restrict__ outF,
                                 __bf16* __restrict__ outB,
                                 int M, int N, int K, int relu) {
    __shared__ __align__(32) __bf16 As[2][128 * 32];
    __shared__ __align__(32) __bf16 Bs[2][128 * 32];

    const int tid    = threadIdx.x;
    const int lane   = tid & 31;
    const int waveId = tid >> 5;
    const int wm     = waveId & 3;       // 0..3  (M dir)
    const int wn     = waveId >> 2;      // 0..1  (N dir)
    const int mWave  = wm * 32;
    const int nWave  = wn * 64;
    const int mBase  = blockIdx.y * 128;
    const int nBase  = blockIdx.x * 128;
    const int half   = lane >> 4;        // K-half selector
    const int lrow   = lane & 15;

    v8f acc[2][4] = {};

    // stage tiles for a given k0 into buffer `buf` (each thread: 2 A + 2 B vec8s)
    auto stage = [&](int k0, int buf) {
        #pragma unroll
        for (int it = 0; it < 2; ++it) {
            int i   = tid + it * 256;    // 0..511
            int row = i >> 2;            // 0..127
            int cv  = i & 3;
            const __bf16* gA = A + (size_t)(mBase + row) * K + k0 + cv * 8;
            const __bf16* gB = W + (size_t)(nBase + row) * K + k0 + cv * 8;
            __bf16* lA = &As[buf][row * 32 + cv * 8];
            __bf16* lB = &Bs[buf][row * 32 + cv * 8];
#if USE_ASYNC_LDS
            __builtin_amdgcn_global_load_async_to_lds_b128(
                (v4i_as1*)gA, (v4i_as3*)lA, 0, 0);
            __builtin_amdgcn_global_load_async_to_lds_b128(
                (v4i_as1*)gB, (v4i_as3*)lB, 0, 0);
#else
            *(uint4*)lA = *(const uint4*)gA;
            *(uint4*)lB = *(const uint4*)gB;
#endif
        }
    };

    // prologue: fill buffer 0
    stage(0, 0);
    wait_async_lds();
    __syncthreads();

    for (int k0 = 0; k0 < K; k0 += 32) {
        const int cur = (k0 >> 5) & 1;
        // issue async stage of next K tile into the other buffer
        if (k0 + 32 < K) stage(k0 + 32, cur ^ 1);

        // fragments from current buffer
        v16bf afrag[2], bfrag[4];
        #pragma unroll
        for (int i = 0; i < 2; ++i)
            afrag[i] = *(const v16bf*)(&As[cur][(mWave + i * 16 + lrow) * 32 + half * 16]);
        #pragma unroll
        for (int j = 0; j < 4; ++j)
            bfrag[j] = *(const v16bf*)(&Bs[cur][(nWave + j * 16 + lrow) * 32 + half * 16]);

        #pragma unroll
        for (int i = 0; i < 2; ++i)
            #pragma unroll
            for (int j = 0; j < 4; ++j)
                acc[i][j] = __builtin_amdgcn_wmma_f32_16x16x32_bf16(
                    false, afrag[i], false, bfrag[j],
                    (short)0, acc[i][j], false, false);

        wait_async_lds();     // own async stage done (each wave waits for its own)
        __syncthreads();      // all waves: stage complete + fragment reads done
    }

    // ---- epilogue: C/D layout: VGPR r -> M = r + 8*half, N = lane&15 ----
    #pragma unroll
    for (int i = 0; i < 2; ++i) {
        #pragma unroll
        for (int j = 0; j < 4; ++j) {
            #pragma unroll
            for (int r = 0; r < 8; ++r) {
                int m = mBase + mWave + i * 16 + r + 8 * half;
                int n = nBase + nWave + j * 16 + lrow;
                float v = acc[i][j][r];
                if (bias) v += bias[n];
                if (relu) v = fmaxf(v, 0.0f);
                size_t idx = (size_t)m * N + n;
                if (outF) outF[idx] = v;
                if (outB) outB[idx] = f2bf(v);
            }
        }
    }
}

// ---------------- reparameterization: z = mu + eps*exp(0.5*lv) ----------------
__global__ void reparam_kernel(const float* __restrict__ mu,
                               const float* __restrict__ lv,
                               const float* __restrict__ eps,
                               float* __restrict__ zF,
                               __bf16* __restrict__ zB, int n) {
    int i = blockIdx.x * blockDim.x + threadIdx.x;
    if (i < n) {
        float z = mu[i] + eps[i] * __expf(0.5f * lv[i]);
        zF[i] = z;
        zB[i] = f2bf(z);
    }
}

// ---------------- sequential RLS scan: one workgroup per class ----------------
// W: [C, L] f32 (mutable copy), P: [C, L, L] f32 (mutable copy).
// Per-class scans are independent; P (1 MB/class, 10 MB total) stays L2-resident.
__global__ void rls_kernel(const float* __restrict__ z,
                           const float* __restrict__ label,
                           float* __restrict__ W,
                           float* __restrict__ P,
                           const int* __restrict__ mode) {
    if (*mode != 1) return;
    __shared__ __align__(16) float xs[Ln];
    __shared__ __align__(16) float Pxs[Ln];
    __shared__ float red[256];
    __shared__ float denom_s, e_s;

    const int c   = blockIdx.x;
    const int tid = threadIdx.x;
    float* Pc = P + (size_t)c * Ln * Ln;
    float* Wc = W + (size_t)c * Ln;

    for (int t = 0; t < Bn; ++t) {
        // load x = z[t]
        for (int i = tid; i < Ln; i += 256) xs[i] = z[(size_t)t * Ln + i];
        __syncthreads();

        // Px = P @ x : each thread owns rows tid and tid+256
        const float4* x4 = (const float4*)xs;
        #pragma unroll
        for (int rr = 0; rr < 2; ++rr) {
            int i = tid + rr * 256;
            const float4* p4 = (const float4*)(Pc + (size_t)i * Ln);
            float s = 0.0f;
            #pragma unroll 4
            for (int j = 0; j < Ln / 4; ++j) {
                float4 p = p4[j], x = x4[j];
                s += p.x * x.x + p.y * x.y + p.z * x.z + p.w * x.w;
            }
            Pxs[i] = s;
        }
        __syncthreads();

        // denom = 1 + Px.x ; e = W.x - y
        float pd = 0.0f, pe = 0.0f;
        for (int i = tid; i < Ln; i += 256) {
            pd += Pxs[i] * xs[i];
            pe += Wc[i] * xs[i];
        }
        red[tid] = pd; __syncthreads();
        for (int s = 128; s > 0; s >>= 1) {
            if (tid < s) red[tid] += red[tid + s];
            __syncthreads();
        }
        if (tid == 0) denom_s = 1.0f + red[0];
        __syncthreads();
        red[tid] = pe; __syncthreads();
        for (int s = 128; s > 0; s >>= 1) {
            if (tid < s) red[tid] += red[tid + s];
            __syncthreads();
        }
        if (tid == 0) e_s = red[0] - label[(size_t)t * Cn + c];
        __syncthreads();

        const float inv = 1.0f / denom_s;
        const float e   = e_s;

        // W -= e * k ; k = Px/denom
        for (int i = tid; i < Ln; i += 256) Wc[i] -= e * Pxs[i] * inv;

        // P -= k (outer) Px  (same row partition as the matvec -> no hazards)
        const float4* Px4 = (const float4*)Pxs;
        #pragma unroll
        for (int rr = 0; rr < 2; ++rr) {
            int i = tid + rr * 256;
            float ki = Pxs[i] * inv;
            float4* p4 = (float4*)(Pc + (size_t)i * Ln);
            #pragma unroll 4
            for (int j = 0; j < Ln / 4; ++j) {
                float4 p = p4[j], q = Px4[j];
                p.x -= ki * q.x; p.y -= ki * q.y; p.z -= ki * q.z; p.w -= ki * q.w;
                p4[j] = p;
            }
        }
        __syncthreads();
    }
}

// ---------------- prediction = z @ W_new^T ----------------
__global__ void pred_kernel(const float* __restrict__ z,
                            const float* __restrict__ W,
                            float* __restrict__ out) {
    __shared__ float xs[Ln];
    int b = blockIdx.x;
    for (int i = threadIdx.x; i < Ln; i += blockDim.x) xs[i] = z[(size_t)b * Ln + i];
    __syncthreads();
    if (threadIdx.x < Cn) {
        int c = threadIdx.x;
        float s = 0.0f;
        for (int l = 0; l < Ln; ++l) s += xs[l] * W[(size_t)c * Ln + l];
        out[(size_t)b * Cn + c] = s;
    }
}

// ---------------- launcher ----------------
extern "C" void kernel_launch(void* const* d_in, const int* in_sizes, int n_in,
                              void* d_out, int out_size, void* d_ws, size_t ws_size,
                              hipStream_t stream) {
    (void)in_sizes; (void)n_in; (void)out_size; (void)ws_size;

    const float* data  = (const float*)d_in[0];
    const float* label = (const float*)d_in[1];
    const float* eps   = (const float*)d_in[2];
    const float* W_e1  = (const float*)d_in[3];
    const float* b_e1  = (const float*)d_in[4];
    const float* W_e2  = (const float*)d_in[5];
    const float* b_e2  = (const float*)d_in[6];
    const float* W_mu  = (const float*)d_in[7];
    const float* b_mu  = (const float*)d_in[8];
    const float* W_lv  = (const float*)d_in[9];
    const float* b_lv  = (const float*)d_in[10];
    const float* W_d1  = (const float*)d_in[11];
    const float* b_d1  = (const float*)d_in[12];
    const float* W_d2  = (const float*)d_in[13];
    const float* b_d2  = (const float*)d_in[14];
    const float* W_out = (const float*)d_in[15];
    const float* P_rls = (const float*)d_in[16];
    const int*   mode  = (const int*)d_in[17];

    // ---- output regions (f32, concatenated in return order) ----
    float* out  = (float*)d_out;
    float* encO = out;
    float* muO  = out + (size_t)Bn * Ln;
    float* lvO  = out + 2 * (size_t)Bn * Ln;
    float* zO   = out + 3 * (size_t)Bn * Ln;
    float* decO = out + 4 * (size_t)Bn * Ln;
    float* prdO = out + 4 * (size_t)Bn * Ln + (size_t)Bn * Dn;

    // ---- workspace carve-up ----
    char* ws = (char*)d_ws;
    size_t off = 0;
    auto carve = [&](size_t bytes) -> char* {
        char* p = ws + off;
        off = (off + bytes + 255) & ~(size_t)255;
        return p;
    };
    __bf16* data_bf = (__bf16*)carve((size_t)Bn * Dn * 2);
    __bf16* We1_bf  = (__bf16*)carve((size_t)Hn * Dn * 2);
    __bf16* h1_bf   = (__bf16*)carve((size_t)Bn * Hn * 2);
    __bf16* We2_bf  = (__bf16*)carve((size_t)Ln * Hn * 2);
    __bf16* enc_bf  = (__bf16*)carve((size_t)Bn * Ln * 2);
    __bf16* Wmu_bf  = (__bf16*)carve((size_t)Ln * Ln * 2);
    __bf16* Wlv_bf  = (__bf16*)carve((size_t)Ln * Ln * 2);
    __bf16* z_bf    = (__bf16*)carve((size_t)Bn * Ln * 2);
    __bf16* Wd1_bf  = (__bf16*)carve((size_t)Hn * Ln * 2);
    __bf16* h2_bf   = (__bf16*)carve((size_t)Bn * Hn * 2);
    __bf16* Wd2_bf  = (__bf16*)carve((size_t)Dn * Hn * 2);
    float*  W_ws    = (float*)carve((size_t)Cn * Ln * 4);
    float*  P_ws    = (float*)carve((size_t)Cn * Ln * Ln * 4);

    auto cvt = [&](const float* s, __bf16* d, size_t n) {
        cvt_bf16_kernel<<<dim3((unsigned)((n + 255) / 256)), dim3(256), 0, stream>>>(s, d, (int)n);
    };
    auto gemm = [&](const __bf16* A, const __bf16* Wt, const float* bias,
                    float* oF, __bf16* oB, int M, int N, int K, int relu) {
        gemm_bf16_kernel<<<dim3(N / 128, M / 128), dim3(256), 0, stream>>>(
            A, Wt, bias, oF, oB, M, N, K, relu);
    };

    // ---- stage bf16 operands ----
    cvt(data, data_bf, (size_t)Bn * Dn);
    cvt(W_e1, We1_bf,  (size_t)Hn * Dn);
    cvt(W_e2, We2_bf,  (size_t)Ln * Hn);
    cvt(W_mu, Wmu_bf,  (size_t)Ln * Ln);
    cvt(W_lv, Wlv_bf,  (size_t)Ln * Ln);
    cvt(W_d1, Wd1_bf,  (size_t)Hn * Ln);
    cvt(W_d2, Wd2_bf,  (size_t)Dn * Hn);

    // ---- encoder ----
    gemm(data_bf, We1_bf, b_e1, nullptr, h1_bf, Bn, Hn, Dn, 1);          // h1 = relu(...)
    gemm(h1_bf,  We2_bf, b_e2, encO,    enc_bf, Bn, Ln, Hn, 1);          // encoded
    gemm(enc_bf, Wmu_bf, b_mu, muO,     nullptr, Bn, Ln, Ln, 0);         // mu
    gemm(enc_bf, Wlv_bf, b_lv, lvO,     nullptr, Bn, Ln, Ln, 0);         // log_var

    // ---- z = mu + eps * exp(0.5*lv) ----
    reparam_kernel<<<dim3((Bn * Ln + 255) / 256), dim3(256), 0, stream>>>(
        muO, lvO, eps, zO, z_bf, Bn * Ln);

    // ---- decoder ----
    gemm(z_bf,  Wd1_bf, b_d1, nullptr, h2_bf, Bn, Hn, Ln, 1);            // h2 = relu(...)
    gemm(h2_bf, Wd2_bf, b_d2, decO,    nullptr, Bn, Dn, Hn, 0);          // decoded

    // ---- RLS: mutable copies of W_out / P_rls, then per-class sequential scan ----
    (void)hipMemcpyAsync(W_ws, W_out, (size_t)Cn * Ln * 4, hipMemcpyDeviceToDevice, stream);
    (void)hipMemcpyAsync(P_ws, P_rls, (size_t)Cn * Ln * Ln * 4, hipMemcpyDeviceToDevice, stream);
    rls_kernel<<<dim3(Cn), dim3(256), 0, stream>>>(zO, label, W_ws, P_ws, mode);

    // ---- prediction = z @ W_new^T ----
    pred_kernel<<<dim3(Bn), dim3(128), 0, stream>>>(zO, W_ws, prdO);
}
